// SelfAttention_56745107915042
// MI455X (gfx1250) — matrix operand order repaired and checked by reading, not compiled, not run on hardware
//
#include <hip/hip_runtime.h>
#include <hip/hip_bf16.h>

// ---------------- CDNA5 WMMA types ----------------
typedef __attribute__((ext_vector_type(16))) __bf16 v16bf;
typedef __attribute__((ext_vector_type(8)))  __bf16 v8bf;
typedef __attribute__((ext_vector_type(8)))  float  v8f;
typedef __attribute__((ext_vector_type(4)))  float  v4f;

#define DD 1024
#define SS 2048
#define BB 8

// ---------------------------------------------------------------------------
// CDNA5 async global->LDS copy (ASYNCcnt path, ISA 08_async_tensor §4).
// ---------------------------------------------------------------------------
__device__ __forceinline__ void async_copy_b128(void* lds_ptr, const void* gptr) {
    unsigned int       loff = (unsigned int)(unsigned long long)(uintptr_t)lds_ptr;
    unsigned long long gad  = (unsigned long long)(uintptr_t)gptr;
    asm volatile("global_load_async_to_lds_b128 %0, %1, off"
                 :: "v"(loff), "v"(gad) : "memory");
}
__device__ __forceinline__ void async_wait0() {
    asm volatile("s_wait_asynccnt 0x0" ::: "memory");
}

// A-fragment (16x32 bf16) load: 16 contiguous K values per lane
__device__ __forceinline__ v16bf load_afrag(const __bf16* p) {
    v16bf a;
    ((v8bf*)&a)[0] = *(const v8bf*)(p);
    ((v8bf*)&a)[1] = *(const v8bf*)(p + 8);
    return a;
}
// B-fragment (32x16 bf16) load: 16 contiguous K values per lane
__device__ __forceinline__ v16bf load_bfrag(const __bf16* p) {
    v16bf b;
    ((v8bf*)&b)[0] = *(const v8bf*)(p);
    ((v8bf*)&b)[1] = *(const v8bf*)(p + 8);
    return b;
}
#define WMMA_BF16(a, b, c) \
    __builtin_amdgcn_wmma_f32_16x16x32_bf16(false, (a), false, (b), (short)0, (c), false, false)

// ---------------------------------------------------------------------------
// Kernel 1: convert x (f32) -> bf16, vectorized
// ---------------------------------------------------------------------------
__global__ __launch_bounds__(256) void cvt_x_kernel(const float* __restrict__ x,
                                                    __bf16* __restrict__ xb, int n) {
    int i = (blockIdx.x * 256 + threadIdx.x) * 8;
    if (i < n) {
        v4f f0 = *(const v4f*)(x + i);
        v4f f1 = *(const v4f*)(x + i + 4);
        v8bf o;
        for (int j = 0; j < 4; ++j) { o[j] = (__bf16)f0[j]; o[4 + j] = (__bf16)f1[j]; }
        *(v8bf*)(xb + i) = o;
    }
}

// ---------------------------------------------------------------------------
// Kernel 2: transpose-convert W (f32 [D,D]) -> Wt bf16 [D,D] (Wt[n][d]=W[d][n])
// ---------------------------------------------------------------------------
__global__ __launch_bounds__(256) void cvt_wt_kernel(const float* __restrict__ Wq,
                                                     const float* __restrict__ Wk,
                                                     const float* __restrict__ Wv,
                                                     __bf16* __restrict__ Wqt,
                                                     __bf16* __restrict__ Wkt,
                                                     __bf16* __restrict__ Wvt) {
    __shared__ float tile[32][33];
    int z = blockIdx.z;
    const float* W  = (z == 0) ? Wq  : (z == 1) ? Wk  : Wv;
    __bf16*      Wt = (z == 0) ? Wqt : (z == 1) ? Wkt : Wvt;
    int n0 = blockIdx.x * 32, d0 = blockIdx.y * 32;
    int tx = threadIdx.x & 31, tyb = threadIdx.x >> 5;
    for (int ty = tyb; ty < 32; ty += 8)
        tile[ty][tx] = W[(size_t)(d0 + ty) * DD + n0 + tx];
    __syncthreads();
    for (int ty = tyb; ty < 32; ty += 8)
        Wt[(size_t)(n0 + ty) * DD + d0 + tx] = (__bf16)tile[tx][ty];
}

// ---------------------------------------------------------------------------
// Kernel 3: QKV projections. Workgroup (8 waves) = 16 rows x 512 cols.
// x-strip async-staged to LDS; each wave computes a 16x64 slab (4 n-tiles).
// Fragment sets are software double-buffered: set for k+32 is issued before
// the WMMAs consuming set k, so loadcnt waits retire early.
// ---------------------------------------------------------------------------
__global__ __launch_bounds__(256) void qkv_gemm_kernel(
    const __bf16* __restrict__ xb,
    const __bf16* __restrict__ Wqt, const __bf16* __restrict__ Wkt,
    const __bf16* __restrict__ Wvt,
    const float* __restrict__ bq, const float* __restrict__ bk,
    const float* __restrict__ bv,
    __bf16* __restrict__ Qb, __bf16* __restrict__ Kb, __bf16* __restrict__ Vt) {
    __shared__ __bf16 xs[16 * DD];     // 32KB A strip
    int tid  = threadIdx.x;
    int wave = tid >> 5;
    int lane = tid & 31;
    int lhalf = lane >> 4;
    int lrow  = lane & 15;
    int z = blockIdx.z;
    const __bf16* Wt   = (z == 0) ? Wqt : (z == 1) ? Wkt : Wvt;
    const float*  bias = (z == 0) ? bq  : (z == 1) ? bk  : bv;

    int m0  = blockIdx.y * 16;
    int n0w = (blockIdx.x * 8 + wave) * 64;      // wave's 64-col slab

    // async-stage the contiguous 16-row x strip into LDS
    {
        const __bf16* src = xb + (size_t)m0 * DD;
        for (int i = tid * 8; i < 16 * DD; i += 256 * 8)
            async_copy_b128(&xs[i], &src[i]);
        async_wait0();
    }
    __syncthreads();

    const __bf16* aptr  = xs + lrow * DD + lhalf * 16;
    const __bf16* bptr0 = Wt + (size_t)(n0w + 0 * 16 + lrow) * DD + lhalf * 16;
    const __bf16* bptr1 = Wt + (size_t)(n0w + 1 * 16 + lrow) * DD + lhalf * 16;
    const __bf16* bptr2 = Wt + (size_t)(n0w + 2 * 16 + lrow) * DD + lhalf * 16;
    const __bf16* bptr3 = Wt + (size_t)(n0w + 3 * 16 + lrow) * DD + lhalf * 16;

    v8f c0 = {}, c1 = {}, c2 = {}, c3 = {};

    // ---- software-pipelined K loop (double-buffered fragment sets) ----
    v16bf aA  = load_afrag(aptr);
    v16bf bA0 = load_bfrag(bptr0), bA1 = load_bfrag(bptr1);
    v16bf bA2 = load_bfrag(bptr2), bA3 = load_bfrag(bptr3);
    #pragma unroll 4
    for (int k = 0; k < DD; k += 64) {
        // issue set B (k+32) before consuming set A (k)
        v16bf aB  = load_afrag(aptr + k + 32);
        v16bf bB0 = load_bfrag(bptr0 + k + 32), bB1 = load_bfrag(bptr1 + k + 32);
        v16bf bB2 = load_bfrag(bptr2 + k + 32), bB3 = load_bfrag(bptr3 + k + 32);
        c0 = WMMA_BF16(aA, bA0, c0);
        c1 = WMMA_BF16(aA, bA1, c1);
        c2 = WMMA_BF16(aA, bA2, c2);
        c3 = WMMA_BF16(aA, bA3, c3);
        // issue set A (k+64) before consuming set B (k+32)
        if (k + 64 < DD) {
            aA  = load_afrag(aptr + k + 64);
            bA0 = load_bfrag(bptr0 + k + 64); bA1 = load_bfrag(bptr1 + k + 64);
            bA2 = load_bfrag(bptr2 + k + 64); bA3 = load_bfrag(bptr3 + k + 64);
        }
        c0 = WMMA_BF16(aB, bB0, c0);
        c1 = WMMA_BF16(aB, bB1, c1);
        c2 = WMMA_BF16(aB, bB2, c2);
        c3 = WMMA_BF16(aB, bB3, c3);
    }

    v8f cs[4] = {c0, c1, c2, c3};
    for (int nt = 0; nt < 4; ++nt) {
        int n = n0w + nt * 16 + lrow;            // C/D column N = lrow
        float bval = bias[n];
        if (z < 2) {
            __bf16* outp = (z == 0) ? Qb : Kb;
            for (int r = 0; r < 8; ++r) {        // C/D: M = r + 8*lhalf
                int m = r + 8 * lhalf;
                outp[(size_t)(m0 + m) * DD + n] = (__bf16)(cs[nt][r] + bval);
            }
        } else {
            for (int r = 0; r < 8; ++r) {        // Vt[b][n][s]
                int m = r + 8 * lhalf;
                int g = m0 + m;
                int bb = g >> 11;
                int s  = g & 2047;
                Vt[((size_t)bb * DD + n) * SS + s] = (__bf16)(cs[nt][r] + bval);
            }
        }
    }
}

// ---------------------------------------------------------------------------
// Kernel 4: flash-style attention. Workgroup (8 waves) = one 16-query tile.
// 16 iterations over 128-key blocks: scores via WMMA (wave w -> 16 keys),
// online softmax in LDS, then each wave accumulates its 128-wide output slice
// with 32 WMMAs (P from LDS, Vt from global, B-fragments batch-loaded).
// ---------------------------------------------------------------------------
__global__ __launch_bounds__(256) void attn_kernel(
    const __bf16* __restrict__ Qb, const __bf16* __restrict__ Kb,
    const __bf16* __restrict__ Vt, const float* __restrict__ x,
    const float* __restrict__ gamma, float* __restrict__ out) {
    __shared__ __bf16 qs[16 * DD];       // 32KB Q tile
    __shared__ float  sc[16 * 128];      // 8KB  scores (f32)
    __shared__ __bf16 ps[16 * 128];      // 4KB  probabilities (bf16)
    __shared__ float  pmax[16 * 16];
    __shared__ float  psum[16 * 16];
    __shared__ float  mrun[16], lrun[16], alpha_s[16];

    int tid   = threadIdx.x;
    int wave  = tid >> 5;
    int lane  = tid & 31;
    int lhalf = lane >> 4;
    int lrow  = lane & 15;
    int row   = tid & 15;                // softmax: row this thread helps with
    int cg    = tid >> 4;                // softmax: 8-key column group

    int qt = blockIdx.x;                 // 0..1023
    int b  = qt >> 7;                    // batch
    int s0 = (qt & 127) << 4;            // query row base within batch
    size_t g0 = (size_t)b * SS + s0;     // global row base

    // async-stage Q tile (16 contiguous rows) into LDS
    {
        const __bf16* qsrc = Qb + g0 * DD;
        for (int i = tid * 8; i < 16 * DD; i += 256 * 8)
            async_copy_b128(&qs[i], &qsrc[i]);
        async_wait0();
    }
    if (tid < 16) { mrun[tid] = -1e30f; lrun[tid] = 0.f; }

    v8f o[8] = {};                       // 16 x 128 f32 output accumulator
    __syncthreads();

    for (int kb = 0; kb < SS; kb += 128) {
        // ---- scores: wave w computes keys [kb + w*16, +16) ----
        int key0 = kb + wave * 16;
        const __bf16* aptr = qs + lrow * DD + lhalf * 16;
        const __bf16* bptr = Kb + ((size_t)b * SS + key0 + lrow) * DD + lhalf * 16;
        if (kb + 128 < SS)  // prefetch next K block (global_prefetch_b8)
            __builtin_prefetch(Kb + ((size_t)b * SS + key0 + 128 + lrow) * DD, 0, 1);
        v8f csc = {};
        #pragma unroll 8
        for (int k = 0; k < DD; k += 32)
            csc = WMMA_BF16(load_afrag(aptr + k), load_bfrag(bptr + k), csc);
        for (int r = 0; r < 8; ++r)
            sc[(r + 8 * lhalf) * 128 + wave * 16 + lrow] = csc[r];
        __syncthreads();

        // ---- online softmax ----
        const float* srow = sc + row * 128 + cg * 8;
        float pm = -1e30f;
        for (int j = 0; j < 8; ++j) pm = fmaxf(pm, srow[j]);
        pmax[row * 16 + cg] = pm;
        __syncthreads();
        if (tid < 16) {
            float m_old = mrun[tid];
            float m_new = m_old;
            for (int j = 0; j < 16; ++j) m_new = fmaxf(m_new, pmax[tid * 16 + j]);
            float a_ = __expf(m_old - m_new);
            alpha_s[tid] = a_;
            mrun[tid]    = m_new;
            lrun[tid]   *= a_;
        }
        __syncthreads();

        float mnew = mrun[row];
        float lpart = 0.f;
        for (int j = 0; j < 8; ++j) {
            float p = __expf(srow[j] - mnew);
            ps[row * 128 + cg * 8 + j] = (__bf16)p;
            lpart += p;
        }
        psum[row * 16 + cg] = lpart;
        __syncthreads();
        if (tid < 16) {
            float t = 0.f;
            for (int j = 0; j < 16; ++j) t += psum[tid * 16 + j];
            lrun[tid] += t;
        }

        // ---- rescale accumulator rows by alpha ----
        float al[8];
        for (int r = 0; r < 8; ++r) al[r] = alpha_s[r + 8 * lhalf];
        for (int nt = 0; nt < 8; ++nt)
            for (int r = 0; r < 8; ++r) o[nt][r] *= al[r];

        // ---- PV: wave w owns output dims [w*128, +128) ----
        const __bf16* vbase = Vt + ((size_t)b * DD + wave * 128 + lrow) * SS
                              + kb + lhalf * 16;
        for (int kk = 0; kk < 4; ++kk) {          // 32 keys per WMMA K-step
            v16bf a = load_afrag(ps + lrow * 128 + kk * 32 + lhalf * 16);
            v16bf bmv[8];
            #pragma unroll
            for (int nt = 0; nt < 8; ++nt)        // batch the 8 B-fragment loads
                bmv[nt] = load_bfrag(vbase + (size_t)nt * 16 * SS + kk * 32);
            #pragma unroll
            for (int nt = 0; nt < 8; ++nt)
                o[nt] = WMMA_BF16(a, bmv[nt], o[nt]);
        }
        __syncthreads();
    }

    // ---- epilogue: out = gamma * O / l + x ----
    float gm = gamma[0];
    for (int nt = 0; nt < 8; ++nt) {
        int n = wave * 128 + nt * 16 + lrow;
        for (int r = 0; r < 8; ++r) {
            int m = r + 8 * lhalf;
            float linv = 1.f / lrun[m];
            size_t idx = (g0 + m) * DD + n;
            out[idx] = gm * o[nt][r] * linv + x[idx];
        }
    }
}

// ---------------------------------------------------------------------------
// Launch: cvt_x -> cvt_wt -> qkv_gemm -> attn   (all in-order on `stream`)
// Workspace layout (~134 MB): xb | Wqt | Wkt | Wvt | Qb | Kb | Vt
// ---------------------------------------------------------------------------
extern "C" void kernel_launch(void* const* d_in, const int* in_sizes, int n_in,
                              void* d_out, int out_size, void* d_ws, size_t ws_size,
                              hipStream_t stream) {
    const float* x     = (const float*)d_in[0];
    const float* Wq    = (const float*)d_in[1];
    const float* bq    = (const float*)d_in[2];
    const float* Wk    = (const float*)d_in[3];
    const float* bk    = (const float*)d_in[4];
    const float* Wv    = (const float*)d_in[5];
    const float* bv    = (const float*)d_in[6];
    const float* gamma = (const float*)d_in[7];
    float* out = (float*)d_out;

    char* ws = (char*)d_ws;
    const size_t XB   = 0;
    const size_t WQT  = XB + (size_t)16384 * 1024 * 2;
    const size_t WKT  = WQT + (size_t)1024 * 1024 * 2;
    const size_t WVT  = WKT + (size_t)1024 * 1024 * 2;
    const size_t QBO  = WVT + (size_t)1024 * 1024 * 2;
    const size_t KBO  = QBO + (size_t)16384 * 1024 * 2;
    const size_t VTO  = KBO + (size_t)16384 * 1024 * 2;

    __bf16* xb  = (__bf16*)(ws + XB);
    __bf16* Wqt = (__bf16*)(ws + WQT);
    __bf16* Wkt = (__bf16*)(ws + WKT);
    __bf16* Wvt = (__bf16*)(ws + WVT);
    __bf16* Qb  = (__bf16*)(ws + QBO);
    __bf16* Kb  = (__bf16*)(ws + KBO);
    __bf16* Vt  = (__bf16*)(ws + VTO);

    cvt_x_kernel<<<8192, 256, 0, stream>>>(x, xb, 16384 * 1024);
    cvt_wt_kernel<<<dim3(32, 32, 3), 256, 0, stream>>>(Wq, Wk, Wv, Wqt, Wkt, Wvt);
    qkv_gemm_kernel<<<dim3(2, 1024, 3), 256, 0, stream>>>(xb, Wqt, Wkt, Wvt,
                                                          bq, bk, bv, Qb, Kb, Vt);
    attn_kernel<<<1024, 256, 0, stream>>>(Qb, Kb, Vt, x, gamma, out);
}
